// CausalSelfAttention_88029649699100
// MI455X (gfx1250) — matrix operand order repaired and checked by reading, not compiled, not run on hardware
//
#include <hip/hip_runtime.h>
#include <hip/hip_bf16.h>

typedef __attribute__((ext_vector_type(16))) _Float16 v16h;
typedef __attribute__((ext_vector_type(8)))  _Float16 v8h;
typedef __attribute__((ext_vector_type(8)))  float    v8f;

#define T_SEQ 4096
#define EMB   1024
#define NH    16
#define HD    64

static __device__ __forceinline__ v8f zero8() {
  v8f z;
#pragma unroll
  for (int i = 0; i < 8; ++i) z[i] = 0.0f;
  return z;
}

static __device__ __forceinline__ v8f wmma_f16(v16h a, v16h b, v8f c) {
  // (neg_a, A, neg_b, B, c_mod, C, reuse_a, reuse_b)
  return __builtin_amdgcn_wmma_f32_16x16x32_f16(false, a, false, b, (short)0, c, false, false);
}

// --- CDNA5 async global->LDS copy (ASYNCcnt-tracked, bypasses VGPRs) -------
static __device__ __forceinline__ void async_g2l_b128(void* lds_ptr, const void* gptr) {
  const unsigned loff = (unsigned)(unsigned long long)lds_ptr;   // low 32b of flat = LDS offset
  const unsigned long long ga = (unsigned long long)gptr;
  asm volatile("global_load_async_to_lds_b128 %0, %1, off" :: "v"(loff), "v"(ga) : "memory");
}
static __device__ __forceinline__ void wait_async0() {
  asm volatile("s_wait_asynccnt 0x0" ::: "memory");
}
static __device__ __forceinline__ void wait_async4() {
  asm volatile("s_wait_asynccnt 0x4" ::: "memory");
}

// A fragment 16x32 f16 from row-major LDS tile. lanes 0-15 -> M=lane,
// K={0..15}; lanes 16-31 -> K={16..31}. One contiguous 32B run -> 2x ds_load_b128.
static __device__ __forceinline__ v16h frag_a(const _Float16* base, int ld, int c, int lane) {
  const int m = lane & 15, hi = lane >> 4;
  const _Float16* row = base + m * ld + c * 32 + 16 * hi;
  v8h lo = *(const v8h*)row;
  v8h hh = *(const v8h*)(row + 8);
  v16h a;
#pragma unroll
  for (int i = 0; i < 8; ++i) { a[i] = lo[i]; a[8 + i] = hh[i]; }
  return a;
}

// B fragment 32x16 where (k,n) is stored transposed: base[n*ld + k]. lane->N,
// K = c*32 + 16*hi + {0..15} (this code's order, as in frag_a): one contiguous 32B run -> 2x ds_load_b128.
static __device__ __forceinline__ v16h frag_bt(const _Float16* base, int ld, int n_off, int c, int lane) {
  const int n = lane & 15, hi = lane >> 4;
  const _Float16* row = base + (n_off + n) * ld + c * 32 + 16 * hi;
  v8h lo = *(const v8h*)row;
  v8h hh = *(const v8h*)(row + 8);
  v16h b;
#pragma unroll
  for (int i = 0; i < 8; ++i) { b[i] = lo[i]; b[8 + i] = hh[i]; }
  return b;
}

// B fragment from fragment-major storage: element (k,n) at base[((k>>4)*ldn + n)*16 + (k&15)].
// Lane's 16 halves are contiguous -> 2x ds_load_b128, zero shuffles.
static __device__ __forceinline__ v16h frag_bf(const _Float16* base, int ldn, int n_off, int c, int lane) {
  const int n = lane & 15, hi = lane >> 4;
  const _Float16* p = base + (size_t)((c * 2 + hi) * ldn + n_off + n) * 16;
  v8h lo = *(const v8h*)p;
  v8h hh = *(const v8h*)(p + 8);
  v16h b;
#pragma unroll
  for (int i = 0; i < 8; ++i) { b[i] = lo[i]; b[8 + i] = hh[i]; }
  return b;
}

// ---------------------------------------------------------------- fp32 -> fp16 (plain)
__global__ void cvt_f32_to_f16(const float* __restrict__ src, _Float16* __restrict__ dst, int n) {
  int i = blockIdx.x * 256 + threadIdx.x;
  if (i < n) dst[i] = (_Float16)src[i];
}

// fp32 -> fp16 weight, rearranged into tile-major fragment-major layout:
// tile (kt=k/32, nt=n/128) contiguous 4096-half block; in-tile:
// addr = ((k%32 / 16)*128 + n%128)*16 + (k%16). Staging = verbatim async b128.
__global__ void cvt_w_swz(const float* __restrict__ src, _Float16* __restrict__ dst, int N) {
  int i = blockIdx.x * 256 + threadIdx.x;
  if (i >= EMB * N) return;
  const int k = i / N, n = i % N;
  const int kt = k >> 5, nt = n >> 7, kk = k & 31, nn = n & 127;
  const size_t off = ((size_t)(kt * (N >> 7) + nt)) * 4096 +
                     ((size_t)((kk >> 4) * 128 + nn)) * 16 + (kk & 15);
  dst[off] = (_Float16)src[i];
}

// ---------------------------------------------------------------- QKV GEMM
__global__ __launch_bounds__(256) void qkv_gemm_kernel(
    const _Float16* __restrict__ Ah, const _Float16* __restrict__ Bsw,
    const float* __restrict__ bias, _Float16* __restrict__ qkvh) {
  __shared__ __align__(16) _Float16 At[2][128 * 32];
  __shared__ __align__(16) _Float16 Bt[2][32 * 128];
  const int tid = threadIdx.x;
  const int lane = tid & 31, wid = tid >> 5;
  const int wm = wid >> 1, wn = wid & 1;
  const int nt_blk = blockIdx.x;                 // 0..23
  const int n0 = nt_blk * 128, m0 = blockIdx.y * 128;
  v8f acc[2][4];
#pragma unroll
  for (int i = 0; i < 2; ++i)
#pragma unroll
    for (int j = 0; j < 4; ++j) acc[i][j] = zero8();

  auto stage = [&](int buf, int k0) {            // 4 async instr / wave
#pragma unroll
    for (int it = 0; it < 2; ++it) {
      const int c = tid + it * 256;              // 512 x 16B chunks
      const int r = c >> 2, off = (c & 3) * 8;
      async_g2l_b128(&At[buf][c * 8], &Ah[(size_t)(m0 + r) * EMB + k0 + off]);
    }
#pragma unroll
    for (int it = 0; it < 2; ++it) {
      const int c = tid + it * 256;
      async_g2l_b128(&Bt[buf][c * 8],
                     &Bsw[((size_t)((k0 >> 5) * 24 + nt_blk)) * 4096 + (size_t)c * 8]);
    }
  };

  stage(0, 0);
  for (int ks = 0; ks < EMB / 32; ++ks) {
    const int cur = ks & 1;
    const bool more = (ks + 1) < EMB / 32;
    if (more) stage(cur ^ 1, (ks + 1) * 32);
    if (more) wait_async4(); else wait_async0();
    __syncthreads();
    v16h bfr[4];
#pragma unroll
    for (int nj = 0; nj < 4; ++nj) bfr[nj] = frag_bf(Bt[cur], 128, wn * 64 + nj * 16, 0, lane);
#pragma unroll
    for (int mi = 0; mi < 2; ++mi) {
      v16h afr = frag_a(At[cur] + (wm * 32 + mi * 16) * 32, 32, 0, lane);
#pragma unroll
      for (int nj = 0; nj < 4; ++nj) acc[mi][nj] = wmma_f16(afr, bfr[nj], acc[mi][nj]);
    }
    __syncthreads();                             // protect buffer refilled next iter
  }

  const int nloc = lane & 15, hi = lane >> 4;
#pragma unroll
  for (int mi = 0; mi < 2; ++mi)
#pragma unroll
    for (int nj = 0; nj < 4; ++nj) {
      const int gn = n0 + wn * 64 + nj * 16 + nloc;
      const float bv = bias[gn];
      const int which = gn >> 10, rem = gn & 1023;   // uniform per nj
      const int head = rem >> 6, dd = rem & 63;
      if (which == 2) {
        // V: [head][ktile][fragment-major 64x64 tile]
        _Float16* vbase = qkvh + (size_t)(2 * NH + head) * T_SEQ * HD;
#pragma unroll
        for (int r = 0; r < 8; ++r) {
          const int gm = m0 + wm * 32 + mi * 16 + r + 8 * hi;
          vbase[(size_t)(gm >> 6) * 4096 +
                ((size_t)(((gm & 63) >> 4) * 64 + dd)) * 16 + (gm & 15)] =
              (_Float16)(acc[mi][nj][r] + bv);
        }
      } else {
        const float scl = (which == 0) ? 0.125f : 1.0f;   // fold 1/sqrt(D) into Q
        _Float16* outp = qkvh + (size_t)(which * NH + head) * T_SEQ * HD + dd;
#pragma unroll
        for (int r = 0; r < 8; ++r) {
          const int gm = m0 + wm * 32 + mi * 16 + r + 8 * hi;
          outp[(size_t)gm * HD] = (_Float16)((acc[mi][nj][r] + bv) * scl);
        }
      }
    }
}

// ---------------------------------------------------------------- Flash attention
// grid (T/64, H), block 128 (4 waves). Q pre-scaled; V tiles fragment-major in global.
__global__ __launch_bounds__(128) void attn_kernel(const _Float16* __restrict__ qkvh,
                                                   _Float16* __restrict__ Oh) {
  __shared__ __align__(16) _Float16 Qt[64 * 64];
  __shared__ __align__(16) _Float16 Kt[64 * 64];
  __shared__ __align__(16) _Float16 Vt[64 * 64];
  __shared__ float    Sf[4 * 16 * 64];
  __shared__ _Float16 Pt[4 * 16 * 64];
  __shared__ float    alphaw[64];
  __shared__ float    lw[64];

  const int head = blockIdx.y;
  const int qb = blockIdx.x;
  const int q0 = qb * 64;
  const int tid = threadIdx.x, lane = tid & 31, w = tid >> 5;
  const _Float16* Qg = qkvh + (size_t)(0 * NH + head) * T_SEQ * HD;
  const _Float16* Kg = qkvh + (size_t)(1 * NH + head) * T_SEQ * HD;
  const _Float16* Vg = qkvh + (size_t)(2 * NH + head) * T_SEQ * HD;  // fragment-major tiles

  // async-stage Q tile (verbatim row-major)
#pragma unroll
  for (int it = 0; it < 4; ++it) {
    const int c = tid + it * 128;                // 512 x 16B chunks
    const int r = c >> 3, off = (c & 7) * 8;
    async_g2l_b128(&Qt[c * 8], &Qg[(size_t)(q0 + r) * HD + off]);
  }
  wait_async0();
  __syncthreads();

  v16h aq0 = frag_a(Qt + (w * 16) * HD, HD, 0, lane);
  v16h aq1 = frag_a(Qt + (w * 16) * HD, HD, 1, lane);
  v8f o[4];
#pragma unroll
  for (int nj = 0; nj < 4; ++nj) o[nj] = zero8();

  const int srow = lane >> 1, hh = lane & 1;
  const int qi = q0 + w * 16 + srow;
  float mprev = -INFINITY, lsum = 0.0f;

  for (int kt = 0; kt <= qb; ++kt) {
    // async-stage K (row-major) + V (fragment-major, verbatim) tiles
#pragma unroll
    for (int it = 0; it < 4; ++it) {
      const int c = tid + it * 128;
      const int r = c >> 3, off = (c & 7) * 8;
      async_g2l_b128(&Kt[c * 8], &Kg[(size_t)(kt * 64 + r) * HD + off]);
      async_g2l_b128(&Vt[c * 8], &Vg[(size_t)kt * 4096 + (size_t)c * 8]);
    }
    wait_async0();
    __syncthreads();

    // S = Q * K^T (contraction over d, contiguous in Kt rows)
    {
      const int nloc = lane & 15, hi = lane >> 4;
#pragma unroll
      for (int nj = 0; nj < 4; ++nj) {
        v8f s = zero8();
        v16h b0 = frag_bt(Kt, HD, nj * 16, 0, lane);
        s = wmma_f16(aq0, b0, s);
        v16h b1 = frag_bt(Kt, HD, nj * 16, 1, lane);
        s = wmma_f16(aq1, b1, s);
#pragma unroll
        for (int r = 0; r < 8; ++r)
          Sf[(w * 16 + r + 8 * hi) * 64 + nj * 16 + nloc] = s[r];
      }
    }
    __syncthreads();

    // online softmax: 2 lanes per row
    {
      int nvalid = qi - kt * 64 + 1; if (nvalid > 64) nvalid = 64;
      const float* Srow = &Sf[(w * 16 + srow) * 64];
      _Float16* Prow = &Pt[(w * 16 + srow) * 64];
      float mloc = -INFINITY;
#pragma unroll
      for (int i = 0; i < 32; ++i) {
        const int col = hh * 32 + i;
        const float sv = Srow[col];
        if (col < nvalid) mloc = fmaxf(mloc, sv);
      }
      mloc = fmaxf(mloc, __shfl_xor(mloc, 1, 32));
      const float mnew = fmaxf(mprev, mloc);
      float psum = 0.0f;
#pragma unroll
      for (int i = 0; i < 32; ++i) {
        const int col = hh * 32 + i;
        const float p = (col < nvalid) ? __expf(Srow[col] - mnew) : 0.0f;
        Prow[col] = (_Float16)p;
        psum += p;
      }
      psum += __shfl_xor(psum, 1, 32);
      const float alpha = __expf(mprev - mnew);
      lsum = lsum * alpha + psum;
      mprev = mnew;
      if (hh == 0) alphaw[w * 16 + srow] = alpha;
    }
    __syncthreads();

    // O = O*alpha + P*V
    {
      const int hi = lane >> 4;
      float al[8];
#pragma unroll
      for (int r = 0; r < 8; ++r) al[r] = alphaw[w * 16 + r + 8 * hi];
#pragma unroll
      for (int nj = 0; nj < 4; ++nj)
#pragma unroll
        for (int r = 0; r < 8; ++r) o[nj][r] *= al[r];
      v16h ap0 = frag_a(Pt + (w * 16) * 64, 64, 0, lane);
      v16h ap1 = frag_a(Pt + (w * 16) * 64, 64, 1, lane);
#pragma unroll
      for (int nj = 0; nj < 4; ++nj) {
        v16h bv0 = frag_bf(Vt, HD, nj * 16, 0, lane);
        o[nj] = wmma_f16(ap0, bv0, o[nj]);
        v16h bv1 = frag_bf(Vt, HD, nj * 16, 1, lane);
        o[nj] = wmma_f16(ap1, bv1, o[nj]);
      }
    }
    __syncthreads();
  }

  if (hh == 0) lw[w * 16 + srow] = lsum;
  __syncthreads();
  {
    const int nloc = lane & 15, hi = lane >> 4;
    float li[8];
#pragma unroll
    for (int r = 0; r < 8; ++r) li[r] = 1.0f / lw[w * 16 + r + 8 * hi];
#pragma unroll
    for (int nj = 0; nj < 4; ++nj) {
      const int col = head * HD + nj * 16 + nloc;
#pragma unroll
      for (int r = 0; r < 8; ++r) {
        const int row = q0 + w * 16 + r + 8 * hi;
        Oh[(size_t)row * EMB + col] = (_Float16)(o[nj][r] * li[r]);
      }
    }
  }
}

// ---------------------------------------------------------------- output projection
__global__ __launch_bounds__(256) void out_gemm_kernel(
    const _Float16* __restrict__ Ah, const _Float16* __restrict__ Bsw,
    const float* __restrict__ bias, float* __restrict__ out) {
  __shared__ __align__(16) _Float16 At[2][128 * 32];
  __shared__ __align__(16) _Float16 Bt[2][32 * 128];
  const int tid = threadIdx.x;
  const int lane = tid & 31, wid = tid >> 5;
  const int wm = wid >> 1, wn = wid & 1;
  const int nt_blk = blockIdx.x;                 // 0..7
  const int n0 = nt_blk * 128, m0 = blockIdx.y * 128;
  v8f acc[2][4];
#pragma unroll
  for (int i = 0; i < 2; ++i)
#pragma unroll
    for (int j = 0; j < 4; ++j) acc[i][j] = zero8();

  auto stage = [&](int buf, int k0) {
#pragma unroll
    for (int it = 0; it < 2; ++it) {
      const int c = tid + it * 256;
      const int r = c >> 2, off = (c & 3) * 8;
      async_g2l_b128(&At[buf][c * 8], &Ah[(size_t)(m0 + r) * EMB + k0 + off]);
    }
#pragma unroll
    for (int it = 0; it < 2; ++it) {
      const int c = tid + it * 256;
      async_g2l_b128(&Bt[buf][c * 8],
                     &Bsw[((size_t)((k0 >> 5) * 8 + nt_blk)) * 4096 + (size_t)c * 8]);
    }
  };

  stage(0, 0);
  for (int ks = 0; ks < EMB / 32; ++ks) {
    const int cur = ks & 1;
    const bool more = (ks + 1) < EMB / 32;
    if (more) stage(cur ^ 1, (ks + 1) * 32);
    if (more) wait_async4(); else wait_async0();
    __syncthreads();
    v16h bfr[4];
#pragma unroll
    for (int nj = 0; nj < 4; ++nj) bfr[nj] = frag_bf(Bt[cur], 128, wn * 64 + nj * 16, 0, lane);
#pragma unroll
    for (int mi = 0; mi < 2; ++mi) {
      v16h afr = frag_a(At[cur] + (wm * 32 + mi * 16) * 32, 32, 0, lane);
#pragma unroll
      for (int nj = 0; nj < 4; ++nj) acc[mi][nj] = wmma_f16(afr, bfr[nj], acc[mi][nj]);
    }
    __syncthreads();
  }

  const int nloc = lane & 15, hi = lane >> 4;
#pragma unroll
  for (int mi = 0; mi < 2; ++mi)
#pragma unroll
    for (int nj = 0; nj < 4; ++nj) {
      const int gn = n0 + wn * 64 + nj * 16 + nloc;
      const float bv = bias[gn];
#pragma unroll
      for (int r = 0; r < 8; ++r) {
        const int gm = m0 + wm * 32 + mi * 16 + r + 8 * hi;
        out[(size_t)gm * EMB + gn] = acc[mi][nj][r] + bv;
      }
    }
}

// ---------------------------------------------------------------- launch
extern "C" void kernel_launch(void* const* d_in, const int* in_sizes, int n_in,
                              void* d_out, int out_size, void* d_ws, size_t ws_size,
                              hipStream_t stream) {
  const float* x      = (const float*)d_in[0];
  const float* w_qkv  = (const float*)d_in[1];
  const float* b_qkv  = (const float*)d_in[2];
  const float* w_out  = (const float*)d_in[3];
  const float* b_out  = (const float*)d_in[4];
  float* out = (float*)d_out;

  _Float16* ws = (_Float16*)d_ws;
  const size_t nX   = (size_t)T_SEQ * EMB;           // 4,194,304
  const size_t nWq  = (size_t)EMB * 3 * EMB;         // 3,145,728
  const size_t nWo  = (size_t)EMB * EMB;             // 1,048,576
  const size_t nQKV = (size_t)3 * NH * T_SEQ * HD;   // 12,582,912
  _Float16* Xh    = ws;
  _Float16* Wqkvh = Xh + nX;          // tile-major fragment-major
  _Float16* Wouth = Wqkvh + nWq;      // tile-major fragment-major
  _Float16* QKVh  = Wouth + nWo;      // Q,K row-major [H][T][D]; V fragment-major tiles
  _Float16* Ohh   = QKVh + nQKV;      // [4096,1024] fp16

  cvt_f32_to_f16<<<(int)((nX + 255) / 256), 256, 0, stream>>>(x, Xh, (int)nX);
  cvt_w_swz<<<(int)((nWq + 255) / 256), 256, 0, stream>>>(w_qkv, Wqkvh, 3 * EMB);
  cvt_w_swz<<<(int)((nWo + 255) / 256), 256, 0, stream>>>(w_out, Wouth, EMB);

  qkv_gemm_kernel<<<dim3(3 * EMB / 128, T_SEQ / 128), 256, 0, stream>>>(Xh, Wqkvh, b_qkv, QKVh);
  attn_kernel<<<dim3(T_SEQ / 64, NH), 128, 0, stream>>>(QKVh, Ohh);
  out_gemm_kernel<<<dim3(EMB / 128, T_SEQ / 128), 256, 0, stream>>>(Ohh, Wouth, b_out, out);
}
